// MultiHeadAttention_42494406426817
// MI455X (gfx1250) — matrix operand order repaired and checked by reading, not compiled, not run on hardware
//
#include <hip/hip_runtime.h>

#define EMBED 1024
#define HEADS 16
#define HDIM  64
#define BATCH 2
#define SEQ   2048
#define NQKV  (3 * EMBED)

typedef unsigned int  u32x4 __attribute__((ext_vector_type(4)));
typedef unsigned int  u32x2 __attribute__((ext_vector_type(2)));
typedef int           i32x8 __attribute__((ext_vector_type(8)));
typedef int           i32x4 __attribute__((ext_vector_type(4)));
typedef float         f32x4 __attribute__((ext_vector_type(4)));
typedef float         v8f   __attribute__((ext_vector_type(8)));
typedef __bf16        v16bf __attribute__((ext_vector_type(16)));

union Frag { v16bf v; u32x4 q[2]; };

__device__ __forceinline__ unsigned short f2bf(float f) {
  unsigned int u = __float_as_uint(f);
  u += 0x7fffu + ((u >> 16) & 1u);        // round-to-nearest-even
  return (unsigned short)(u >> 16);
}
__device__ __forceinline__ unsigned int pk2(float a, float b) {
  return (unsigned int)f2bf(a) | ((unsigned int)f2bf(b) << 16);
}
__device__ __forceinline__ v8f wmma_bf16(Frag a, Frag b, v8f c) {
  return __builtin_amdgcn_wmma_f32_16x16x32_bf16(false, a.v, false, b.v,
                                                 (short)0, c, false, false);
}
// generic shared pointer -> 32-bit LDS byte offset (aperture low bits)
__device__ __forceinline__ unsigned lds_off(const void* p) {
  return (unsigned)(unsigned long long)(uintptr_t)p;
}
// Tensor Data Mover: 2D tile (rows x cols bf16), row stride in elements,
// LDS rows padded by 16B every 128B (matches [.][72] ushort arrays).
__device__ __forceinline__ void tdm_load_2d_bf16(
    unsigned ldsaddr, const unsigned short* gsrc,
    int cols, int rows, long long stride_elems) {
  u32x4 g0;
  unsigned long long ga = (unsigned long long)(uintptr_t)gsrc;
  g0[0] = 1u;                                  // count=1, load, user
  g0[1] = ldsaddr;                             // lds_addr
  g0[2] = (unsigned)ga;                        // global_addr[31:0]
  g0[3] = (unsigned)(ga >> 32) | (2u << 30);   // global_addr[56:32] | type=2
  i32x8 g1;
  g1[0] = (1 << 16)      // data_size = 2 bytes
        | (1 << 20)      // pad_enable
        | (4 << 22)      // pad_interval: every 32 DWORDs (128B row)
        | (3 << 25);     // pad_amount: 4 DWORDs (16B)
  g1[1] = (cols & 0xffff) << 16;                       // tensor_dim0 lo16
  g1[2] = ((cols >> 16) & 0xffff) | ((rows & 0xffff) << 16);  // dim0 hi | dim1 lo
  g1[3] = ((rows >> 16) & 0xffff) | ((cols & 0xffff) << 16);  // dim1 hi | tile_dim0
  g1[4] = rows & 0xffff;                               // tile_dim1 (tile_dim2=0)
  g1[5] = (int)(stride_elems & 0xffffffffll);          // tensor_dim0_stride lo32
  g1[6] = (int)((stride_elems >> 32) & 0xffff);        // stride hi16 | dim1_stride lo=0
  g1[7] = 0;
  i32x4 gz4; gz4[0] = 0; gz4[1] = 0; gz4[2] = 0; gz4[3] = 0;
  i32x8 gz8;
#pragma unroll
  for (int i = 0; i < 8; ++i) gz8[i] = 0;
  __builtin_amdgcn_tensor_load_to_lds(g0, g1, gz4, gz4, gz8, 0);
}
__device__ __forceinline__ void async_copy_b128(unsigned ldsaddr,
                                                const unsigned short* gsrc) {
  asm volatile("global_load_async_to_lds_b128 %0, %1, off"
               :: "v"(ldsaddr), "v"(gsrc) : "memory");
}

// ---------------------------------------------------------------------------
// Kernel 1: QKV projection  C[4096,3072] = x[4096,1024] @ w_in + b_in
// 128x128 tile / WG, 8 waves (4 in M x 2 in N), wave tile 32x64, BK=32.
// Writes Q (scaled by 1/sqrt(D)), K, V as bf16 in [B,H,S,D].
// ---------------------------------------------------------------------------
__global__ __launch_bounds__(256) void qkv_wmma(
    const float* __restrict__ x, const float* __restrict__ w_in,
    const float* __restrict__ b_in,
    unsigned short* __restrict__ Qo, unsigned short* __restrict__ Ko,
    unsigned short* __restrict__ Vo) {
  __shared__ unsigned short As[128][40];   // [m][k], row pad -> 80B stride
  __shared__ unsigned short Bs[128][40];   // [n][k] (B transposed)
  const int tid  = threadIdx.x;
  const int lane = tid & 31, wid = tid >> 5;
  const int wm = wid & 3, wn = wid >> 2;
  const int m0 = blockIdx.y * 128, n0 = blockIdx.x * 128;
  const int half = lane >> 4, l16 = lane & 15;
  const int kloA = half * 8, kloB = half * 8;

  v8f acc[2][4];
#pragma unroll
  for (int i = 0; i < 2; ++i)
#pragma unroll
    for (int j = 0; j < 4; ++j)
#pragma unroll
      for (int r = 0; r < 8; ++r) acc[i][j][r] = 0.f;

  for (int k0 = 0; k0 < EMBED; k0 += 32) {
    // stage A: 128x32 fp32 -> bf16
#pragma unroll
    for (int i = 0; i < 4; ++i) {
      int g = tid + i * 256;
      int row = g >> 3, col = (g & 7) * 4;
      const float* ap = x + (size_t)(m0 + row) * EMBED + k0 + col;
      f32x4 a4 = *(const f32x4*)ap;
      if (k0 + 32 < EMBED) __builtin_prefetch(ap + 32, 0, 1);
      u32x2 p; p[0] = pk2(a4[0], a4[1]); p[1] = pk2(a4[2], a4[3]);
      *(u32x2*)&As[row][col] = p;
    }
    // stage B transposed: Bs[n][k] = w_in[k0+k][n0+n]
#pragma unroll
    for (int i = 0; i < 4; ++i) {
      int g = tid + i * 256;
      int k = g >> 5, n = (g & 31) * 4;
      f32x4 b4 = *(const f32x4*)(w_in + (size_t)(k0 + k) * NQKV + n0 + n);
      Bs[n + 0][k] = f2bf(b4[0]); Bs[n + 1][k] = f2bf(b4[1]);
      Bs[n + 2][k] = f2bf(b4[2]); Bs[n + 3][k] = f2bf(b4[3]);
    }
    __syncthreads();
    Frag a[2], b[4];
#pragma unroll
    for (int sm = 0; sm < 2; ++sm) {
      int row = wm * 32 + sm * 16 + l16;
      a[sm].q[0] = *(const u32x4*)&As[row][kloA];
      a[sm].q[1] = *(const u32x4*)&As[row][kloA + 16];
    }
#pragma unroll
    for (int sn = 0; sn < 4; ++sn) {
      int col = wn * 64 + sn * 16 + l16;
      b[sn].q[0] = *(const u32x4*)&Bs[col][kloB];
      b[sn].q[1] = *(const u32x4*)&Bs[col][kloB + 16];
    }
#pragma unroll
    for (int sm = 0; sm < 2; ++sm)
#pragma unroll
      for (int sn = 0; sn < 4; ++sn)
        acc[sm][sn] = wmma_bf16(a[sm], b[sn], acc[sm][sn]);
    __syncthreads();
  }
  // epilogue: bias, split into Q/K/V [B,H,S,D] bf16 (Q pre-scaled by 1/8)
#pragma unroll
  for (int sm = 0; sm < 2; ++sm)
#pragma unroll
    for (int sn = 0; sn < 4; ++sn) {
      int gn = n0 + wn * 64 + sn * 16 + l16;
      float bias = b_in[gn];
      int which = gn >> 10;
      int e = gn & 1023, h = e >> 6, d = e & 63;
      unsigned short* dst = which == 0 ? Qo : (which == 1 ? Ko : Vo);
      float mul = which == 0 ? 0.125f : 1.0f;
#pragma unroll
      for (int r = 0; r < 8; ++r) {
        int gm = m0 + wm * 32 + sm * 16 + r + half * 8;
        int bb = gm >> 11, s = gm & (SEQ - 1);
        size_t o = (((size_t)bb * HEADS + h) * SEQ + s) * HDIM + d;
        dst[o] = f2bf((acc[sm][sn][r] + bias) * mul);
      }
    }
}

// ---------------------------------------------------------------------------
// Kernel 2: flash attention. One WG = 64 query rows of one (b,h); 4 waves,
// each wave owns 16 query rows. KV streamed in 64-row blocks; K block staged
// by the Tensor Data Mover (with HW padding), V staged+transposed by lanes.
// ---------------------------------------------------------------------------
__global__ __launch_bounds__(128) void attn_wmma(
    const unsigned short* __restrict__ Qg, const unsigned short* __restrict__ Kg,
    const unsigned short* __restrict__ Vg, unsigned short* __restrict__ Og) {
  __shared__ unsigned short Ks[64][72];      // [s_kv][d]  (QK^T B-layout)
  __shared__ unsigned short Vs[64][72];      // [d][s_kv]  (PV  B-layout)
  __shared__ unsigned short Ps[4][16][72];   // per-wave P, [row][s_kv]
  const int tid  = threadIdx.x;
  const int lane = tid & 31, wid = tid >> 5;
  const int bh = blockIdx.y;
  const int q0 = blockIdx.x * 64;
  const int half = lane >> 4, l16 = lane & 15;
  const int kloA = half * 8, kloB = half * 8;
  const size_t base = (size_t)bh * SEQ;
  const unsigned ks_lds = lds_off(&Ks[0][0]);

  // Q fragments (A-layout), loaded once, D=64 -> two 16x32 k-steps
  Frag qf[2];
  {
    const unsigned short* qp = Qg + (base + q0 + wid * 16 + l16) * HDIM;
    qf[0].q[0] = *(const u32x4*)(qp + kloA);
    qf[0].q[1] = *(const u32x4*)(qp + kloA + 16);
    qf[1].q[0] = *(const u32x4*)(qp + 32 + kloA);
    qf[1].q[1] = *(const u32x4*)(qp + 32 + kloA + 16);
  }
  float mi[8], li[8];
  v8f oacc[4];
#pragma unroll
  for (int r = 0; r < 8; ++r) { mi[r] = -3.0e38f; li[r] = 0.f; }
#pragma unroll
  for (int t = 0; t < 4; ++t)
#pragma unroll
    for (int r = 0; r < 8; ++r) oacc[t][r] = 0.f;

  for (int kv0 = 0; kv0 < SEQ; kv0 += 64) {
    // K block via TDM: 64x64 bf16 tile, LDS rows padded to 144B by HW
    if (wid == 0) {
      tdm_load_2d_bf16(ks_lds, Kg + (base + kv0) * HDIM, HDIM, 64, HDIM);
      __builtin_amdgcn_s_wait_tensorcnt(0);
    }
    // V block: lanes stage + transpose
#pragma unroll
    for (int i = 0; i < 4; ++i) {
      int g = tid + i * 128;
      int row = g >> 3, col = (g & 7) * 8;
      u32x4 raw = *(const u32x4*)(Vg + (base + kv0 + row) * HDIM + col);
#pragma unroll
      for (int j = 0; j < 4; ++j) {
        Vs[col + 2 * j    ][row] = (unsigned short)(raw[j] & 0xffffu);
        Vs[col + 2 * j + 1][row] = (unsigned short)(raw[j] >> 16);
      }
    }
    __syncthreads();
    // S = Q * K^T (16x64 per wave)
    v8f sacc[4];
#pragma unroll
    for (int t = 0; t < 4; ++t)
#pragma unroll
      for (int r = 0; r < 8; ++r) sacc[t][r] = 0.f;
#pragma unroll
    for (int ks = 0; ks < 2; ++ks)
#pragma unroll
      for (int t = 0; t < 4; ++t) {
        Frag kf;
        kf.q[0] = *(const u32x4*)&Ks[t * 16 + l16][ks * 32 + kloB];
        kf.q[1] = *(const u32x4*)&Ks[t * 16 + l16][ks * 32 + kloB + 16];
        sacc[t] = wmma_bf16(qf[ks], kf, sacc[t]);
      }
    // online softmax: row m = r + 8*half lives across the 16 lanes of a half
    float sc[8];
#pragma unroll
    for (int r = 0; r < 8; ++r) {
      float rm = fmaxf(fmaxf(sacc[0][r], sacc[1][r]),
                       fmaxf(sacc[2][r], sacc[3][r]));
#pragma unroll
      for (int msk = 1; msk < 16; msk <<= 1)
        rm = fmaxf(rm, __shfl_xor(rm, msk, 32));
      float mnew = fmaxf(mi[r], rm);
      sc[r] = __expf(mi[r] - mnew);
      float rs = 0.f;
#pragma unroll
      for (int t = 0; t < 4; ++t) {
        float p = __expf(sacc[t][r] - mnew);
        rs += p;
        Ps[wid][r + half * 8][t * 16 + l16] = f2bf(p);
      }
#pragma unroll
      for (int msk = 1; msk < 16; msk <<= 1) rs += __shfl_xor(rs, msk, 32);
      li[r] = li[r] * sc[r] + rs;
      mi[r] = mnew;
    }
#pragma unroll
    for (int t = 0; t < 4; ++t)
#pragma unroll
      for (int r = 0; r < 8; ++r) oacc[t][r] *= sc[r];
    asm volatile("s_wait_dscnt 0" ::: "memory");   // P stores -> P frag loads
    // O += P * V
#pragma unroll
    for (int ks = 0; ks < 2; ++ks) {
      Frag pf;
      pf.q[0] = *(const u32x4*)&Ps[wid][l16][ks * 32 + kloA];
      pf.q[1] = *(const u32x4*)&Ps[wid][l16][ks * 32 + kloA + 16];
#pragma unroll
      for (int t = 0; t < 4; ++t) {
        Frag vf;
        vf.q[0] = *(const u32x4*)&Vs[t * 16 + l16][ks * 32 + kloB];
        vf.q[1] = *(const u32x4*)&Vs[t * 16 + l16][ks * 32 + kloB + 16];
        oacc[t] = wmma_bf16(pf, vf, oacc[t]);
      }
    }
    __syncthreads();
  }
  // normalize and store O as bf16 [B,S,E]
  const int b = bh >> 4, h = bh & 15;
#pragma unroll
  for (int r = 0; r < 8; ++r) {
    float inv = 1.0f / li[r];
    int s = q0 + wid * 16 + r + half * 8;
    size_t orow = ((size_t)b * SEQ + s) * EMBED + h * HDIM;
#pragma unroll
    for (int t = 0; t < 4; ++t)
      Og[orow + t * 16 + l16] = f2bf(oacc[t][r] * inv);
  }
}

// ---------------------------------------------------------------------------
// Kernel 3: output projection  out[4096,1024] = O @ w_out + b_out (fp32 out)
// A tile staged by async global->LDS b128 copies (no VGPR round trip).
// ---------------------------------------------------------------------------
__global__ __launch_bounds__(256) void oproj_wmma(
    const unsigned short* __restrict__ Og, const float* __restrict__ w_out,
    const float* __restrict__ b_out, float* __restrict__ out) {
  __shared__ unsigned short As[128][40];
  __shared__ unsigned short Bs[128][40];
  const int tid  = threadIdx.x;
  const int lane = tid & 31, wid = tid >> 5;
  const int wm = wid & 3, wn = wid >> 2;
  const int m0 = blockIdx.y * 128, n0 = blockIdx.x * 128;
  const int half = lane >> 4, l16 = lane & 15;
  const int kloA = half * 8, kloB = half * 8;

  v8f acc[2][4];
#pragma unroll
  for (int i = 0; i < 2; ++i)
#pragma unroll
    for (int j = 0; j < 4; ++j)
#pragma unroll
      for (int r = 0; r < 8; ++r) acc[i][j][r] = 0.f;

  for (int k0 = 0; k0 < EMBED; k0 += 32) {
    // stage A: async bf16 copy, per-lane padded LDS destinations
#pragma unroll
    for (int i = 0; i < 2; ++i) {
      int g = tid + i * 256;
      int row = g >> 2, col = (g & 3) * 8;
      async_copy_b128(lds_off(&As[row][col]),
                      Og + (size_t)(m0 + row) * EMBED + k0 + col);
    }
    // stage B transposed: Bs[n][k] = w_out[k0+k][n0+n], fp32 -> bf16
#pragma unroll
    for (int i = 0; i < 4; ++i) {
      int g = tid + i * 256;
      int k = g >> 5, n = (g & 31) * 4;
      f32x4 b4 = *(const f32x4*)(w_out + (size_t)(k0 + k) * EMBED + n0 + n);
      Bs[n + 0][k] = f2bf(b4[0]); Bs[n + 1][k] = f2bf(b4[1]);
      Bs[n + 2][k] = f2bf(b4[2]); Bs[n + 3][k] = f2bf(b4[3]);
    }
    asm volatile("s_wait_asynccnt 0x0" ::: "memory");
    __syncthreads();
    Frag a[2], b[4];
#pragma unroll
    for (int sm = 0; sm < 2; ++sm) {
      int row = wm * 32 + sm * 16 + l16;
      a[sm].q[0] = *(const u32x4*)&As[row][kloA];
      a[sm].q[1] = *(const u32x4*)&As[row][kloA + 16];
    }
#pragma unroll
    for (int sn = 0; sn < 4; ++sn) {
      int col = wn * 64 + sn * 16 + l16;
      b[sn].q[0] = *(const u32x4*)&Bs[col][kloB];
      b[sn].q[1] = *(const u32x4*)&Bs[col][kloB + 16];
    }
#pragma unroll
    for (int sm = 0; sm < 2; ++sm)
#pragma unroll
      for (int sn = 0; sn < 4; ++sn)
        acc[sm][sn] = wmma_bf16(a[sm], b[sn], acc[sm][sn]);
    __syncthreads();
  }
#pragma unroll
  for (int sm = 0; sm < 2; ++sm)
#pragma unroll
    for (int sn = 0; sn < 4; ++sn) {
      int gn = n0 + wn * 64 + sn * 16 + l16;
      float bias = b_out[gn];
#pragma unroll
      for (int r = 0; r < 8; ++r) {
        int gm = m0 + wm * 32 + sm * 16 + r + half * 8;
        out[(size_t)gm * EMBED + gn] = acc[sm][sn][r] + bias;
      }
    }
}

// ---------------------------------------------------------------------------
extern "C" void kernel_launch(void* const* d_in, const int* in_sizes, int n_in,
                              void* d_out, int out_size, void* d_ws, size_t ws_size,
                              hipStream_t stream) {
  const float* x     = (const float*)d_in[0];
  const float* w_in  = (const float*)d_in[1];
  const float* b_in  = (const float*)d_in[2];
  const float* w_out = (const float*)d_in[3];
  const float* b_out = (const float*)d_in[4];
  float* out = (float*)d_out;

  const size_t QSZ = (size_t)BATCH * HEADS * SEQ * HDIM;  // 4M bf16 each
  unsigned short* Q = (unsigned short*)d_ws;
  unsigned short* K = Q + QSZ;
  unsigned short* V = K + QSZ;
  unsigned short* O = V + QSZ;                            // total 32 MB

  qkv_wmma<<<dim3(NQKV / 128, (BATCH * SEQ) / 128), 256, 0, stream>>>(
      x, w_in, b_in, Q, K, V);
  attn_wmma<<<dim3(SEQ / 64, BATCH * HEADS), 128, 0, stream>>>(Q, K, V, O);
  oproj_wmma<<<dim3(EMBED / 128, (BATCH * SEQ) / 128), 256, 0, stream>>>(
      O, w_out, b_out, out);
}
